// SOLD2_detector_55336358642330
// MI455X (gfx1250) — compile-verified, matrix operand order checked by reading, not run on hardware
//
#include <hip/hip_runtime.h>
#include <hip/hip_bf16.h>
#include <math.h>

typedef __attribute__((ext_vector_type(16))) _Float16 v16h;
typedef __attribute__((ext_vector_type(8)))  float    v8f;
typedef __attribute__((ext_vector_type(2)))  float    v2f;

#define HH     512
#define WW     512
#define NJ     250
#define NPAIR  31125            // 250*249/2
#define NSAMP  64
#define NBINS  2048
#define CSC    (1.0f/64.0f)     // coord scale for the f16 fallback path only

#if __has_builtin(__builtin_amdgcn_wmma_f32_16x16x4_f32)
#define USE_WMMA_F32 1          // exact-fit K=4 f32 shape, no scaling needed
#else
#define USE_WMMA_F32 0          // fall back to codegen-verified f16 shape
#endif

// ---------------------------------------------------------------- init scratch
__global__ void k_init(int* __restrict__ p) {
    int i = blockIdx.x * blockDim.x + threadIdx.x;
    if (i < 2 * NBINS + 2) p[i] = 0;     // hist, hsum, valid, max20
}

// ------------------------------------------------- histogram (replaces sort)
__global__ void k_hist(const float* __restrict__ heat, int* __restrict__ hist,
                       float* __restrict__ hsum, int* __restrict__ valid) {
    int i      = blockIdx.x * blockDim.x + threadIdx.x;
    int stride = gridDim.x * blockDim.x;          // sized so trip count is uniform
    int lane   = threadIdx.x & 31;
    for (; i < HH * WW; i += stride) {
        float v = heat[i];
        bool ok = v > 0.001f;
        unsigned long long bal = __ballot(ok);
        if (ok) {
            int b = (int)(fminf(fmaxf(v, 0.0f), 0.9999995f) * (float)NBINS);
            atomicAdd(&hist[b], 1);
            atomicAdd(&hsum[b], v);
        }
        if (lane == 0) atomicAdd(valid, (int)__popcll(bal));
    }
}

// ------------------------------------- top-k scan over bins -> max20 scalar
__global__ void k_scan(const int* __restrict__ hist, const float* __restrict__ hsum,
                       const int* __restrict__ valid, float* __restrict__ max20) {
    if (threadIdx.x == 0 && blockIdx.x == 0) {
        int nv = *valid;
        int k  = (int)ceilf((float)nv * 0.2f);
        if (k < 1) k = 1;
        int rem = k; float s = 0.0f;
        for (int b = NBINS - 1; b >= 0 && rem > 0; --b) {
            int c = hist[b];
            if (c <= rem) { s += hsum[b]; rem -= c; }
            else          { s += hsum[b] * ((float)rem / (float)c); rem = 0; }
        }
        float m = s / (float)k;
        *max20 = (m > 1e-12f) ? m : 1.0f;
    }
}

// --------------------------- normalize heatmap + copy junctions to output
__global__ void k_norm(const float* __restrict__ heat, const float* __restrict__ junc,
                       const float* __restrict__ max20, float* __restrict__ heat_r,
                       float* __restrict__ out_h, float* __restrict__ out_j) {
    int i      = blockIdx.x * blockDim.x + threadIdx.x;
    int stride = gridDim.x * blockDim.x;
    float inv  = 1.0f / *max20;
    for (int t = i; t < HH * WW; t += stride) {
        float v = fminf(fmaxf(heat[t] * inv, 0.0f), 1.0f);
        heat_r[t] = v;
        out_h[t]  = v;
    }
    if (i < NJ * 2) out_j[i] = junc[i];
}

// ------------------- triu(k=1) pair index tables + zero line_map output
__global__ void k_pairs(float* __restrict__ lm, int* __restrict__ idx0,
                        int* __restrict__ idx1) {
    int i = blockIdx.x;
    for (int j = threadIdx.x; j < NJ; j += blockDim.x) {
        lm[i * NJ + j] = 0.0f;
        if (j > i) {
            int m = i * (2 * NJ - i - 1) / 2 + (j - i - 1);
            idx0[m] = i; idx1[m] = j;
        }
    }
}

// ---------------------------------------------------------------- keep mask
// Two bilinear forms per (pair m, junction n), as 16x16 WMMA tiles with K=4:
//   A row n  = (h, w, h^2+w^2, 1)                   (junction features)
//   B_dot m  = (dir_h, dir_w, 0, -(s.dir))  -> dot[m,n]      (proj * len)
//   B_cn  m  = (-2 s_h, -2 s_w, 1, |s|^2)   -> ||j[n]-s[m]||^2
// perp^2 = cnorm^2 - dot^2 ; hit iff perp<=3 and 0<=dot<=len.
__global__ void k_keep(const float* __restrict__ junc, const int* __restrict__ idx0,
                       const int* __restrict__ idx1, int* __restrict__ keep) {
    const int  lane = threadIdx.x & 31;
    const int  wave = (blockIdx.x * blockDim.x + threadIdx.x) >> 5;
    const int  p    = lane & 15;                 // pair column / junction row in half
    const bool hi   = lane >= 16;                // carries K=2,3 for f32 shape
    const int  base = wave * 16;
    if (base >= NPAIR) return;                   // uniform per wave

    int  pair   = base + p;
    bool pvalid = pair < NPAIR;
    int  pc     = pvalid ? pair : (NPAIR - 1);
    int  i0 = idx0[pc], i1 = idx1[pc];
    float sh = junc[2 * i0], sw = junc[2 * i0 + 1];
    float eh = junc[2 * i1], ew = junc[2 * i1 + 1];
    float dh = eh - sh, dw = ew - sw;
    float len  = sqrtf(dh * dh + dw * dw);
    float invl = 1.0f / len;
    float dirh = dh * invl, dirw = dw * invl;

#if USE_WMMA_F32
    // B 4x16 f32: VGPR0 = {K0 | K2}, VGPR1 = {K1 | K3} across lane halves
    v2f b_dot, b_cn;
    if (!hi) { b_dot[0] = dirh;       b_dot[1] = dirw;
               b_cn[0]  = -2.0f * sh; b_cn[1]  = -2.0f * sw; }
    else     { b_dot[0] = 0.0f;       b_dot[1] = -(sh * dirh + sw * dirw);
               b_cn[0]  = 1.0f;       b_cn[1]  = sh * sh + sw * sw; }
    const float DSCL = 1.0f, CSCL = 1.0f;
#else
    v16h b_dot = {};
    v16h b_cn  = {};
    if (!hi) {                                   // lanes 16..31 carry the zero K-half
        b_dot[0] = (_Float16)dirh;
        b_dot[1] = (_Float16)dirw;
        b_dot[3] = (_Float16)(-(sh * dirh + sw * dirw) * CSC);
        b_cn[0]  = (_Float16)(-2.0f * sh * CSC);
        b_cn[1]  = (_Float16)(-2.0f * sw * CSC);
        b_cn[2]  = (_Float16)1.0f;
        b_cn[3]  = (_Float16)((sh * sh + sw * sw) * CSC * CSC);
    }
    const float DSCL = 64.0f, CSCL = 4096.0f;
#endif

    int cnt = 0;
    for (int t = 0; t < 16; ++t) {               // 16 junction tiles cover n=0..255
        int   n  = t * 16 + p;
        bool  jv = n < NJ;
        float jh = jv ? junc[2 * n]     : 0.0f;
        float jw = jv ? junc[2 * n + 1] : 0.0f;
#if USE_WMMA_F32
        v2f a;
        if (!hi) { a[0] = jh; a[1] = jw; }
        else     { a[0] = jv ? (jh * jh + jw * jw) : 1.0e9f;  // pad -> cnorm^2 huge
                   a[1] = jv ? 1.0f : 0.0f; }
        v8f z  = {};
        v8f dv = __builtin_amdgcn_wmma_f32_16x16x4_f32(false, a, false, b_dot,
                                                       (short)0, z, false, false);
        v8f cv = __builtin_amdgcn_wmma_f32_16x16x4_f32(false, a, false, b_cn,
                                                       (short)0, z, false, false);
#else
        v16h a = {};
        if (!hi) {
            if (jv) {
                float jhs = jh * CSC, jws = jw * CSC;
                a[0] = (_Float16)jhs; a[1] = (_Float16)jws;
                a[2] = (_Float16)(jhs * jhs + jws * jws);
                a[3] = (_Float16)1.0f;
            } else {
                a[2] = (_Float16)6.0e4f;         // pad -> cnorm^2 huge -> no hit
            }
        }
        v8f z  = {};
        v8f dv = __builtin_amdgcn_wmma_f32_16x16x32_f16(false, a, false, b_dot,
                                                        (short)0, z, false, false);
        v8f cv = __builtin_amdgcn_wmma_f32_16x16x32_f16(false, a, false, b_cn,
                                                        (short)0, z, false, false);
#endif
#pragma unroll
        for (int r = 0; r < 8; ++r) {
            float dot   = dv[r] * DSCL;
            float cn2   = cv[r] * CSCL;
            float perp2 = cn2 - dot * dot;       // cnorm^2 sin^2(theta)
            bool  m = (perp2 <= 9.0f) && (dot >= 0.0f) && (dot <= len);
            cnt += m ? 1 : 0;
        }
    }
    cnt += __shfl_xor(cnt, 16, 32);              // join row halves M<8 / M>=8
    if (lane < 16 && pvalid) keep[pair] = ((cnt - 2) <= 0) ? 1 : 0;  // drop i0,i1 self-hits
}

// ------------------------------------------------ line scores (gather-bound)
__global__ void k_score(const float* __restrict__ junc, const float* __restrict__ heat,
                        const int* __restrict__ idx0, const int* __restrict__ idx1,
                        const int* __restrict__ keep, float* __restrict__ lm) {
    const int lane = threadIdx.x & 31;
    const int pair = (blockIdx.x * blockDim.x + threadIdx.x) >> 5;
    if (pair >= NPAIR) return;                   // uniform per wave
    int i0 = idx0[pair], i1 = idx1[pair];
    float sh = junc[2 * i0], sw = junc[2 * i0 + 1];
    float eh = junc[2 * i1], ew = junc[2 * i1 + 1];
    float dh = sh - eh, dw = sw - ew;
    float len      = sqrtf(dh * dh + dw * dw);
    float norm_len = len * (1.0f / 724.07734f);  // / sqrt(512^2+512^2)
    float dth      = 0.70710678f + 2.0f * norm_len;

    float ssum = 0.0f; int scnt = 0;
#pragma unroll
    for (int si = 0; si < 2; ++si) {             // 64 samples = 32 lanes x 2
        int   s  = lane + si * 32;
        float t  = (float)s * (1.0f / 63.0f);
        float ch = fminf(fmaxf(sh * t + eh * (1.0f - t), 0.0f), 511.0f);
        float cw = fminf(fmaxf(sw * t + ew * (1.0f - t), 0.0f), 511.0f);
        float rh = rintf(ch), rw = rintf(cw);    // round-half-even, matches jnp.round
        __builtin_prefetch(&heat[(int)rh * WW + (int)rw], 0, 1);
        float lmax = 0.0f;
#pragma unroll
        for (int oy = -3; oy <= 3; ++oy)
#pragma unroll
            for (int ox = -3; ox <= 3; ++ox) {
                if (oy * oy + ox * ox > 9) continue;   // circular patch, 29 taps
                float shh = rh + (float)oy, sww = rw + (float)ox;
                float ddh = ch - shh, ddw = cw - sww;
                bool  m  = sqrtf(ddh * ddh + ddw * ddw) < dth;
                int   ph = (int)fminf(fmaxf(shh, 0.0f), 511.0f);
                int   pw = (int)fminf(fmaxf(sww, 0.0f), 511.0f);
                float v  = heat[ph * WW + pw];
                lmax = fmaxf(lmax, m ? v : 0.0f);
            }
        ssum += lmax;
        scnt += (lmax > 0.5f) ? 1 : 0;
    }
#pragma unroll
    for (int off = 16; off >= 1; off >>= 1) {    // wave32 butterfly
        ssum += __shfl_xor(ssum, off, 32);
        scnt += __shfl_xor(scnt, off, 32);
    }
    if (lane == 0) {
        float mean   = ssum * (1.0f / 64.0f);
        float inlier = (float)scnt * (1.0f / 64.0f);
        bool  det = (mean > 0.5f) && (inlier >= 0.99f) && (keep[pair] != 0);
        float v = det ? 1.0f : 0.0f;
        lm[i0 * NJ + i1] = v;
        lm[i1 * NJ + i0] = v;
    }
}

// ---------------------------------------------------------------------------
extern "C" void kernel_launch(void* const* d_in, const int* in_sizes, int n_in,
                              void* d_out, int out_size, void* d_ws, size_t ws_size,
                              hipStream_t stream) {
    (void)in_sizes; (void)n_in; (void)out_size; (void)ws_size;
    const float* junc = (const float*)d_in[0];
    const float* heat = (const float*)d_in[1];

    float* out   = (float*)d_out;
    float* lm    = out;                          // [250*250]
    float* out_j = out + NJ * NJ;                // [500]
    float* out_h = out + NJ * NJ + NJ * 2;       // [512*512]

    float* ws     = (float*)d_ws;
    float* heat_r = ws;                                    // 262144
    int*   hist   = (int*)  (ws + HH * WW);                // 2048
    float* hsum   =         ws + HH * WW + NBINS;          // 2048
    int*   valid  = (int*)  (ws + HH * WW + 2 * NBINS);    // 1
    float* max20  =         ws + HH * WW + 2 * NBINS + 1;  // 1
    int*   idx0   = (int*)  (ws + HH * WW + 2 * NBINS + 2);
    int*   idx1   = idx0 + NPAIR;
    int*   keep   = idx1 + NPAIR;

    k_init <<<17, 256, 0, stream>>>(hist);
    k_hist <<<256, 256, 0, stream>>>(heat, hist, hsum, valid);
    k_scan <<<1, 32, 0, stream>>>(hist, hsum, valid, max20);
    k_norm <<<256, 256, 0, stream>>>(heat, junc, max20, heat_r, out_h, out_j);
    k_pairs<<<NJ, 256, 0, stream>>>(lm, idx0, idx1);

    int nwaves = (NPAIR + 15) / 16;              // one wave per 16-pair tile
    int nkb    = (nwaves + 7) / 8;               // 8 waves (256 thr) per block
    k_keep <<<nkb, 256, 0, stream>>>(junc, idx0, idx1, keep);

    int nsb = (NPAIR * 32 + 255) / 256;          // one wave per pair
    k_score<<<nsb, 256, 0, stream>>>(junc, heat_r, idx0, idx1, keep, lm);
}
